// PathClassifier_19834158973581
// MI455X (gfx1250) — matrix operand order repaired and checked by reading, not compiled, not run on hardware
//
#include <hip/hip_runtime.h>
#include <hip/hip_bf16.h>

// ---------------------------------------------------------------------------
// PathClassifier on MI455X (gfx1250, wave32, WMMA)
//
// score[b,t] = e_t . (W_attn^T init_b)  => q = init @ W_attn  (one small GEMM)
// out        = ctx @ W_lin^T + b_lin    (second small GEMM)
// Both GEMMs run on v_wmma_f32_16x16x32_bf16 (f32 accum). The memory-bound
// part (2 HBM passes over the 201MB encoded_paths) is done by plain coalesced
// f32 kernels; the ctx pass re-reads group data that is L2-resident.
// ---------------------------------------------------------------------------

typedef __bf16 bf16;
typedef __bf16 v16bf __attribute__((ext_vector_type(16)));
typedef __bf16 v8bf  __attribute__((ext_vector_type(8)));
typedef float  v8f   __attribute__((ext_vector_type(8)));

#define D_DIM 1024
#define C_DIM 1024
#define B_GRP 128

// ---------------------------------------------------------------------------
// Kernel: tiled transpose + f32->bf16 convert.  Wt[d][e] = (bf16) W[e][d]
// ---------------------------------------------------------------------------
__global__ void k_transpose_bf16(const float* __restrict__ W,
                                 bf16* __restrict__ Wt, int Dd) {
    __shared__ float tile[32][33];
    int bx = blockIdx.x * 32;   // source col block
    int by = blockIdx.y * 32;   // source row block
    int tx = threadIdx.x;       // 0..31
    int ty = threadIdx.y;       // 0..7
    #pragma unroll
    for (int j = 0; j < 32; j += 8)
        tile[ty + j][tx] = W[(size_t)(by + ty + j) * Dd + bx + tx];
    __syncthreads();
    #pragma unroll
    for (int j = 0; j < 32; j += 8)
        Wt[(size_t)(bx + ty + j) * Dd + by + tx] = (bf16)tile[tx][ty + j];
}

// ---------------------------------------------------------------------------
// Kernel: flat f32->bf16 convert (for W_lin, already in B^T-friendly layout)
// ---------------------------------------------------------------------------
__global__ void k_convert_bf16(const float* __restrict__ src,
                               bf16* __restrict__ dst, int n) {
    int i = blockIdx.x * blockDim.x + threadIdx.x;
    if (i < n) dst[i] = (bf16)src[i];
}

// ---------------------------------------------------------------------------
// Kernel: per-group mean of encoded_paths -> init (bf16, for WMMA A-matrix)
// One block per group, 256 threads, 4 columns per thread. Streams 201MB (HBM).
// ---------------------------------------------------------------------------
__global__ void k_segment_mean(const float* __restrict__ E,
                               const int* __restrict__ len,
                               bf16* __restrict__ initbf) {
    int b = blockIdx.x;
    int tid = threadIdx.x;
    int start = 0;
    for (int i = 0; i < b; ++i) start += len[i];
    int L = len[b];
    float acc0 = 0.f, acc1 = 0.f, acc2 = 0.f, acc3 = 0.f;
    for (int t = 0; t < L; ++t) {
        const float* row = E + (size_t)(start + t) * D_DIM;
        acc0 += row[tid];
        acc1 += row[tid + 256];
        acc2 += row[tid + 512];
        acc3 += row[tid + 768];
    }
    float inv = 1.0f / (float)L;
    bf16* o = initbf + (size_t)b * D_DIM;
    o[tid]       = (bf16)(acc0 * inv);
    o[tid + 256] = (bf16)(acc1 * inv);
    o[tid + 512] = (bf16)(acc2 * inv);
    o[tid + 768] = (bf16)(acc3 * inv);
}

// ---------------------------------------------------------------------------
// Kernel: bf16 WMMA GEMM.  Out[M,N] = A[M,K] @ Bt[N,K]^T (+ bias[N])
//   A  : row-major M x K bf16
//   Bt : row-major N x K bf16 (i.e. B-matrix stored transposed, so each
//        lane's B fragment (16 consecutive K for one N) is contiguous)
// One wave per 16x16 output tile, K-loop in steps of 32.
// Fragment layouts used here (contiguous K, same for A and B; not the ISA table order):
//   A lane l: row m=(l&15); elems 0..15 = K[k0+16h .. k0+16h+16)
//             with h = l>>4
//   B lane l: col n=(l&15); elems 0..15 = K[k0+16h .. k0+16h+16)
//   C lane l: vgpr r -> (M = r + 8*(l>>4), N = (l&15))
// ---------------------------------------------------------------------------
__global__ void k_gemm_wmma_bf16(const bf16* __restrict__ A,
                                 const bf16* __restrict__ Bt,
                                 const float* __restrict__ bias,
                                 float* __restrict__ Out,
                                 int M, int N, int K) {
    int wave = threadIdx.x >> 5;
    int lane = threadIdx.x & 31;
    int wavesPerBlock = blockDim.x >> 5;
    int tile = blockIdx.x * wavesPerBlock + wave;
    int tilesN = N >> 4;
    int tm = tile / tilesN;
    int tn = tile - tm * tilesN;
    if ((tm << 4) >= M) return;

    int l15 = lane & 15;
    int half = lane >> 4;            // 0 or 1
    int m = (tm << 4) + l15;         // A row for this lane
    int n = (tn << 4) + l15;         // B column for this lane

    const bf16* arow = A + (size_t)m * K;
    const bf16* brow = Bt + (size_t)n * K;

    v8f acc = {};
    for (int k0 = 0; k0 < K; k0 += 32) {
        v8bf alo = *(const v8bf*)(arow + k0 + half * 16);
        v8bf ahi = *(const v8bf*)(arow + k0 + half * 16 + 8);
        v16bf a = __builtin_shufflevector(alo, ahi,
                                          0, 1, 2, 3, 4, 5, 6, 7,
                                          8, 9, 10, 11, 12, 13, 14, 15);
        v16bf bfrag = *(const v16bf*)(brow + k0 + half * 16);
        acc = __builtin_amdgcn_wmma_f32_16x16x32_bf16(
            /*neg_a=*/false, a, /*neg_b=*/false, bfrag,
            /*c_mod=*/(short)0, acc, /*reuse_a=*/false, /*reuse_b=*/false);
    }

    float badd = bias ? bias[n] : 0.0f;
    int mbase = (tm << 4) + half * 8;
    int ncol = (tn << 4) + l15;
    #pragma unroll
    for (int r = 0; r < 8; ++r)
        Out[(size_t)(mbase + r) * N + ncol] = acc[r] + badd;
}

// ---------------------------------------------------------------------------
// Kernel: fused per-group attention.  One block (256 thr = 8 waves) per group.
//   pass 1: score[t] = e_t . q_b          (wave-per-row dot, HBM read)
//   pass 2: softmax over the group        (LDS scores)
//   pass 3: ctx = sum_t w_t * e_t         (thread-per-col, L2-resident re-read)
// ---------------------------------------------------------------------------
__global__ void k_attention(const float* __restrict__ E,
                            const int* __restrict__ len,
                            const float* __restrict__ q,
                            bf16* __restrict__ ctxbf) {
    __shared__ float s_sc[512];
    __shared__ float s_red[32];

    int b = blockIdx.x;
    int tid = threadIdx.x;
    int wave = tid >> 5;
    int lane = tid & 31;

    int start = 0;
    for (int i = 0; i < b; ++i) start += len[i];
    int L = len[b];
    const float* qb = q + (size_t)b * D_DIM;

    // ---- pass 1: scores ----
    for (int t = wave; t < L; t += 8) {
        const float* row = E + (size_t)(start + t) * D_DIM;
        float s = 0.f;
        #pragma unroll 8
        for (int j = lane; j < D_DIM; j += 32) s += row[j] * qb[j];
        #pragma unroll
        for (int off = 16; off > 0; off >>= 1) s += __shfl_xor(s, off, 32);
        if (lane == 0) s_sc[t] = s;
    }
    __syncthreads();

    // ---- pass 2: softmax ----
    float m = -3.0e38f;
    for (int t = tid; t < L; t += 256) m = fmaxf(m, s_sc[t]);
    #pragma unroll
    for (int off = 16; off > 0; off >>= 1) m = fmaxf(m, __shfl_xor(m, off, 32));
    if (lane == 0) s_red[wave] = m;
    __syncthreads();
    if (tid == 0) {
        float mm = s_red[0];
        #pragma unroll
        for (int i = 1; i < 8; ++i) mm = fmaxf(mm, s_red[i]);
        s_red[8] = mm;
    }
    __syncthreads();
    float gmax = s_red[8];

    float lsum = 0.f;
    for (int t = tid; t < L; t += 256) {
        float w = __expf(s_sc[t] - gmax);
        s_sc[t] = w;
        lsum += w;
    }
    #pragma unroll
    for (int off = 16; off > 0; off >>= 1) lsum += __shfl_xor(lsum, off, 32);
    if (lane == 0) s_red[16 + wave] = lsum;
    __syncthreads();
    if (tid == 0) {
        float ss = 0.f;
        #pragma unroll
        for (int i = 0; i < 8; ++i) ss += s_red[16 + i];
        s_red[24] = 1.0f / ss;
    }
    __syncthreads();
    float inv = s_red[24];
    for (int t = tid; t < L; t += 256) s_sc[t] *= inv;
    __syncthreads();

    // ---- pass 3: weighted sum (rows now in L2) ----
    float acc0 = 0.f, acc1 = 0.f, acc2 = 0.f, acc3 = 0.f;
    for (int t = 0; t < L; ++t) {
        float w = s_sc[t];
        const float* row = E + (size_t)(start + t) * D_DIM;
        acc0 += w * row[tid];
        acc1 += w * row[tid + 256];
        acc2 += w * row[tid + 512];
        acc3 += w * row[tid + 768];
    }
    bf16* o = ctxbf + (size_t)b * D_DIM;
    o[tid]       = (bf16)acc0;
    o[tid + 256] = (bf16)acc1;
    o[tid + 512] = (bf16)acc2;
    o[tid + 768] = (bf16)acc3;
}

// ---------------------------------------------------------------------------
// Launch
// ---------------------------------------------------------------------------
extern "C" void kernel_launch(void* const* d_in, const int* in_sizes, int n_in,
                              void* d_out, int out_size, void* d_ws, size_t ws_size,
                              hipStream_t stream) {
    const float* encoded = (const float*)d_in[0];   // [n_paths, 1024]
    const int*   lens    = (const int*)d_in[1];     // [128]
    const float* W_attn  = (const float*)d_in[2];   // [1024, 1024]  (e,d)
    const float* W_lin   = (const float*)d_in[3];   // [1024, 1024]  (c,d)
    const float* b_lin   = (const float*)d_in[4];   // [1024]
    float*       out     = (float*)d_out;           // [128, 1024]

    // workspace carving
    char* ws = (char*)d_ws;
    bf16* WtA_bf  = (bf16*)ws;                       ws += (size_t)D_DIM * D_DIM * sizeof(bf16); // W_attn^T bf16
    bf16* Wl_bf   = (bf16*)ws;                       ws += (size_t)C_DIM * D_DIM * sizeof(bf16); // W_lin bf16
    bf16* init_bf = (bf16*)ws;                       ws += (size_t)B_GRP * D_DIM * sizeof(bf16);
    bf16* ctx_bf  = (bf16*)ws;                       ws += (size_t)B_GRP * D_DIM * sizeof(bf16);
    float* qbuf   = (float*)ws;                      ws += (size_t)B_GRP * D_DIM * sizeof(float);

    // 1) weight conversions (independent of data path)
    dim3 tb(32, 8);
    dim3 tg(D_DIM / 32, D_DIM / 32);
    k_transpose_bf16<<<tg, tb, 0, stream>>>(W_attn, WtA_bf, D_DIM);
    k_convert_bf16<<<(C_DIM * D_DIM) / 256, 256, 0, stream>>>(W_lin, Wl_bf, C_DIM * D_DIM);

    // 2) segment mean -> init (bf16)
    k_segment_mean<<<B_GRP, 256, 0, stream>>>(encoded, lens, init_bf);

    // 3) q = init @ W_attn   (WMMA; Bt = W_attn^T so fragments are contiguous)
    //    tiles: (128/16) x (1024/16) = 512; 8 waves/block -> 64 blocks
    k_gemm_wmma_bf16<<<64, 256, 0, stream>>>(init_bf, WtA_bf, nullptr, qbuf,
                                             B_GRP, D_DIM, D_DIM);

    // 4) fused scores/softmax/ctx per group
    k_attention<<<B_GRP, 256, 0, stream>>>(encoded, lens, qbuf, ctx_bf);

    // 5) out = ctx @ W_lin^T + b_lin  (WMMA; W_lin rows are already K-major)
    k_gemm_wmma_bf16<<<64, 256, 0, stream>>>(ctx_bf, Wl_bf, b_lin, out,
                                             B_GRP, C_DIM, D_DIM);
}
